// Darnn_selfattention_59356448031600
// MI455X (gfx1250) — compile-verified, matrix operand order checked by reading, not run on hardware
//
#include <hip/hip_runtime.h>
#include <hip/hip_bf16.h>

typedef __attribute__((ext_vector_type(16))) _Float16 v16h;
typedef __attribute__((ext_vector_type(8)))  _Float16 v8h;
typedef __attribute__((ext_vector_type(8)))  float    v8f;
typedef __attribute__((ext_vector_type(4)))  int      i32x4;

#define BATCH 4096
#define TLEN  45
#define HH    512
#define TT    9     // encoded sequence length

#if defined(__has_builtin)
#if __has_builtin(__builtin_amdgcn_global_load_async_to_lds_b128) && __has_builtin(__builtin_amdgcn_s_wait_asynccnt)
#define USE_ASYNC_LDS 1
#endif
#endif

#define AS1 __attribute__((address_space(1)))
#define AS3 __attribute__((address_space(3)))

// ---------------------------------------------------------------- elementwise
__global__ void k_zero_f16(_Float16* p, long n) {
  for (long i = blockIdx.x * 256L + threadIdx.x; i < n; i += gridDim.x * 256L) p[i] = (_Float16)0.f;
}
__global__ void k_zero_f32(float* p, long n) {
  for (long i = blockIdx.x * 256L + threadIdx.x; i < n; i += gridDim.x * 256L) p[i] = 0.f;
}
__global__ void k_f32_to_f16(const float* s, _Float16* d, long n) {
  for (long i = blockIdx.x * 256L + threadIdx.x; i < n; i += gridDim.x * 256L) d[i] = (_Float16)s[i];
}
// fused LSTM weight: rows=2048 (gates), cols = Kx + 512 ; wih row-stride = sih
__global__ void k_build_w(const float* wih, int sih, int kx, const float* whh, _Float16* d, long n) {
  int kc = kx + HH;
  for (long i = blockIdx.x * 256L + threadIdx.x; i < n; i += gridDim.x * 256L) {
    int j = (int)(i / kc), c = (int)(i % kc);
    float v = (c < kx) ? wih[(long)j * sih + c] : whh[(long)j * HH + (c - kx)];
    d[i] = (_Float16)v;
  }
}
__global__ void k_add_bias(const float* a, const float* b, float* o, int n) {
  int i = blockIdx.x * 256 + threadIdx.x; if (i < n) o[i] = a[i] + b[i];
}
__global__ void k_extract_col(const float* m, int stride, int col, float* o, int n) {
  int i = blockIdx.x * 256 + threadIdx.x; if (i < n) o[i] = m[(long)i * stride + col];
}
__global__ void k_copy_slice(const _Float16* src, long srow, _Float16* dst, long drow, int cols, long n) {
  for (long i = blockIdx.x * 256L + threadIdx.x; i < n; i += gridDim.x * 256L) {
    long b = i / cols; int c = (int)(i % cols);
    dst[b * drow + c] = src[b * srow + c];
  }
}

// ---------------------------------------------------------------- RCNN front-end
__global__ __launch_bounds__(256) void k_rcnn(
    const float* __restrict__ x, const float* __restrict__ w1, const float* __restrict__ b1,
    const float* __restrict__ w2, const float* __restrict__ b2,
    const float* __restrict__ w3, const float* __restrict__ b3,
    _Float16* __restrict__ feat, int stride, int Ls, int chanoff)
{
  __shared__ float sh1[16 * 45];
  __shared__ float sh2[32 * 43];
  __shared__ float sp2[32 * 21];
  __shared__ float sh3[32 * 19];
  const int b = blockIdx.x, tid = threadIdx.x;
  const int L2 = Ls - 2, P2 = L2 / 2, L3 = P2 - 2, P3 = L3 / 2;
  const float* xb = x + (long)b * TLEN * 128;

  for (int i = tid; i < 16 * Ls; i += 256) {
    int c = i / Ls, l = i % Ls;
    const float* xp = xb + (long)(l * stride) * 128;
    const float* wp = w1 + c * 128;
    float s = b1[c];
    for (int d = 0; d < 128; ++d) s += xp[d] * wp[d];
    sh1[c * Ls + l] = s;
  }
  __syncthreads();
  for (int i = tid; i < 32 * L2; i += 256) {
    int c = i / L2, l = i % L2;
    float s = b2[c];
    for (int ic = 0; ic < 16; ++ic)
      for (int k = 0; k < 3; ++k) s += sh1[ic * Ls + l + k] * w2[(c * 16 + ic) * 3 + k];
    sh2[c * L2 + l] = s;
  }
  __syncthreads();
  for (int i = tid; i < 32 * P2; i += 256) {
    int c = i / P2, l = i % P2;
    sp2[c * P2 + l] = fmaxf(sh2[c * L2 + 2 * l], sh2[c * L2 + 2 * l + 1]);
  }
  __syncthreads();
  for (int i = tid; i < 32 * L3; i += 256) {
    int c = i / L3, l = i % L3;
    float s = b3[c];
    for (int ic = 0; ic < 32; ++ic)
      for (int k = 0; k < 3; ++k) s += sp2[ic * P2 + l + k] * w3[(c * 32 + ic) * 3 + k];
    sh3[c * L3 + l] = s;
  }
  __syncthreads();
  for (int i = tid; i < 32 * P3; i += 256) {
    int c = i / P3, l = i % P3;
    float v = fmaxf(sh3[c * L3 + 2 * l], sh3[c * L3 + 2 * l + 1]);
    feat[(long)b * (TT * 128) + (long)(TT - P3 + l) * 128 + chanoff + c] = (_Float16)v;
  }
}

// ---------------------------------------------------------------- WMMA GEMM (NT)
// C[m,n] = alpha * sum_k A[m,k] * W[n,k] ; A: MxK f16 (row stride K), W: NxK f16.
// Block tile 128(M) x 256(N), BK=32, double-buffered LDS; 8 waves, each 64x64.
// MODE 0: f32 out (ldc), MODE 1: f16 out, MODE 2: f16 out transposed C[n*ldc+m]
#define LDT 40
template<int MODE>
__global__ __launch_bounds__(256) void k_gemm(
    const _Float16* __restrict__ A, const _Float16* __restrict__ W,
    void* __restrict__ Cp, int M, int N, int K, int ldc, float alpha)
{
  __shared__ _Float16 sA[2][128 * LDT];
  __shared__ _Float16 sB[2][256 * LDT];
  const int tid = threadIdx.x;
  const int bm = blockIdx.y, bn = blockIdx.x;
  const int wave = tid >> 5, lane = tid & 31;
  const int wm = (wave & 1) * 64, wn = (wave >> 1) * 64;
  const int r = lane & 15, hi = lane >> 4;
  const int kb = hi * 8;

  const _Float16* Ab = A + (long)(bm * 128) * K;
  const _Float16* Wb = W + (long)(bn * 256) * K;

  int arow[2], aoff[2], brow[4], boff[4];
#pragma unroll
  for (int i = 0; i < 2; ++i) { int c = tid + i * 256; arow[i] = c >> 2; aoff[i] = (c & 3) * 8; }
#pragma unroll
  for (int i = 0; i < 4; ++i) { int c = tid + i * 256; brow[i] = c >> 2; boff[i] = (c & 3) * 8; }

  v8f acc[4][4] = {};

  auto compute = [&](int buf) {
    v16h af[4], bf[4];
#pragma unroll
    for (int mt = 0; mt < 4; ++mt) {
      int row = wm + mt * 16 + r;
      union { v16h v; v8h h[2]; } u;
      u.h[0] = *(v8h*)&sA[buf][row * LDT + kb];
      u.h[1] = *(v8h*)&sA[buf][row * LDT + kb + 16];
      af[mt] = u.v;
    }
#pragma unroll
    for (int nt = 0; nt < 4; ++nt) {
      int col = wn + nt * 16 + r;
      union { v16h v; v8h h[2]; } u;
      u.h[0] = *(v8h*)&sB[buf][col * LDT + kb];
      u.h[1] = *(v8h*)&sB[buf][col * LDT + kb + 16];
      bf[nt] = u.v;
    }
#pragma unroll
    for (int mt = 0; mt < 4; ++mt)
#pragma unroll
      for (int nt = 0; nt < 4; ++nt)
        acc[mt][nt] = __builtin_amdgcn_wmma_f32_16x16x32_f16(
            false, af[mt], false, bf[nt], (short)0, acc[mt][nt], false, false);
  };

#ifdef USE_ASYNC_LDS
  auto stage_async = [&](int buf, int k0) {
#pragma unroll
    for (int i = 0; i < 2; ++i)
      __builtin_amdgcn_global_load_async_to_lds_b128(
          (AS1 i32x4*)(Ab + (long)arow[i] * K + k0 + aoff[i]),
          (AS3 i32x4*)&sA[buf][arow[i] * LDT + aoff[i]], 0, 0);
#pragma unroll
    for (int i = 0; i < 4; ++i)
      __builtin_amdgcn_global_load_async_to_lds_b128(
          (AS1 i32x4*)(Wb + (long)brow[i] * K + k0 + boff[i]),
          (AS3 i32x4*)&sB[buf][brow[i] * LDT + boff[i]], 0, 0);
  };
  stage_async(0, 0);
  __builtin_amdgcn_s_wait_asynccnt(0);
  __syncthreads();
  int cur = 0;
  for (int k0 = 0; k0 < K; k0 += 32) {
    if (k0 + 32 < K) stage_async(cur ^ 1, k0 + 32);
    compute(cur);
    __builtin_amdgcn_s_wait_asynccnt(0);
    __syncthreads();
    cur ^= 1;
  }
#else
  v8h ga[2], gb[4];
  auto load_regs = [&](int k0) {
#pragma unroll
    for (int i = 0; i < 2; ++i) ga[i] = *(const v8h*)&Ab[(long)arow[i] * K + k0 + aoff[i]];
#pragma unroll
    for (int i = 0; i < 4; ++i) gb[i] = *(const v8h*)&Wb[(long)brow[i] * K + k0 + boff[i]];
  };
  auto store_stage = [&](int buf) {
#pragma unroll
    for (int i = 0; i < 2; ++i) *(v8h*)&sA[buf][arow[i] * LDT + aoff[i]] = ga[i];
#pragma unroll
    for (int i = 0; i < 4; ++i) *(v8h*)&sB[buf][brow[i] * LDT + boff[i]] = gb[i];
  };
  load_regs(0);
  int cur = 0;
  for (int k0 = 0; k0 < K; k0 += 32) {
    store_stage(cur);
    __syncthreads();
    if (k0 + 32 < K) load_regs(k0 + 32);
    compute(cur);
    cur ^= 1;
  }
#endif

#pragma unroll
  for (int mt = 0; mt < 4; ++mt)
#pragma unroll
    for (int nt = 0; nt < 4; ++nt) {
      int colg = bn * 256 + wn + nt * 16 + r;
#pragma unroll
      for (int e = 0; e < 8; ++e) {
        int rowg = bm * 128 + wm + mt * 16 + hi * 8 + e;
        float v = acc[mt][nt][e] * alpha;
        if (MODE == 0)      ((float*)Cp)[(long)rowg * ldc + colg] = v;
        else if (MODE == 1) ((_Float16*)Cp)[(long)rowg * ldc + colg] = (_Float16)v;
        else                ((_Float16*)Cp)[(long)colg * ldc + rowg] = (_Float16)v;
      }
    }
}

// ---------------------------------------------------------------- LSTM cell
__global__ __launch_bounds__(256) void k_lstm_cell(
    const float* __restrict__ gates, const float* __restrict__ bias, float* __restrict__ c,
    _Float16* __restrict__ Ah, int lda, int hoff,
    _Float16* __restrict__ extra, long estride,
    const float* __restrict__ y, const float* __restrict__ wy, int t)
{
  int idx = blockIdx.x * 256 + threadIdx.x;            // BATCH*512
  int b = idx >> 9, u = idx & 511;
  const float* g = gates + (long)b * 2048;
  float gi = g[u]        + bias[u];
  float gf = g[u + 512]  + bias[u + 512];
  float gg = g[u + 1024] + bias[u + 1024];
  float go = g[u + 1536] + bias[u + 1536];
  if (y) {
    float yv = y[(long)b * TLEN + 4 + 5 * t];
    gi += yv * wy[u]; gf += yv * wy[u + 512]; gg += yv * wy[u + 1024]; go += yv * wy[u + 1536];
  }
  float si = 1.f / (1.f + __expf(-gi));
  float sf = 1.f / (1.f + __expf(-gf));
  float so = 1.f / (1.f + __expf(-go));
  float cn = sf * c[idx] + si * tanhf(gg);
  c[idx] = cn;
  float h = so * tanhf(cn);
  Ah[(long)b * lda + hoff + u] = (_Float16)h;
  if (extra) extra[(long)b * estride + u] = (_Float16)h;
}

// ---------------------------------------------------------------- softmax row
__global__ __launch_bounds__(256) void k_softmax(const float* __restrict__ z, _Float16* __restrict__ beta) {
  __shared__ float red[256];
  const int row = blockIdx.x, tid = threadIdx.x;
  const float* zr = z + (long)row * BATCH;
  float m = -1e30f;
  for (int j = tid; j < BATCH; j += 256) m = fmaxf(m, zr[j]);
  red[tid] = m; __syncthreads();
  for (int s = 128; s > 0; s >>= 1) { if (tid < s) red[tid] = fmaxf(red[tid], red[tid + s]); __syncthreads(); }
  m = red[0]; __syncthreads();
  float sum = 0.f;
  for (int j = tid; j < BATCH; j += 256) sum += __expf(zr[j] - m);
  red[tid] = sum; __syncthreads();
  for (int s = 128; s > 0; s >>= 1) { if (tid < s) red[tid] += red[tid + s]; __syncthreads(); }
  float inv = 1.f / red[0];
  _Float16* br = beta + (long)row * BATCH;
  for (int j = tid; j < BATCH; j += 256) br[j] = (_Float16)(__expf(zr[j] - m) * inv);
}

// ---------------------------------------------------------------- final dot + sigmoid
__global__ __launch_bounds__(256) void k_final(const float* __restrict__ st,
    const float* __restrict__ lnw, const float* __restrict__ lnb, float* __restrict__ out) {
  int wave = threadIdx.x >> 5, lane = threadIdx.x & 31;
  int row = blockIdx.x * 8 + wave;
  const float* s = st + (long)row * HH;
  float sum = 0.f;
  for (int j = lane; j < HH; j += 32) sum += s[j] * lnw[j];
  for (int off = 16; off > 0; off >>= 1) sum += __shfl_down(sum, off, 32);
  if (lane == 0) out[row] = 1.f / (1.f + __expf(-(sum + lnb[0])));
}

// ---------------------------------------------------------------- launcher
extern "C" void kernel_launch(void* const* d_in, const int* in_sizes, int n_in,
                              void* d_out, int out_size, void* d_ws, size_t ws_size,
                              hipStream_t stream) {
  const float* x       = (const float*)d_in[0];
  const float* y       = (const float*)d_in[1];
  const float* rw1     = (const float*)d_in[2];
  const float* rb1     = (const float*)d_in[3];
  const float* rw2     = (const float*)d_in[4];
  const float* rb2     = (const float*)d_in[5];
  const float* rw3     = (const float*)d_in[6];
  const float* rb3     = (const float*)d_in[7];
  const float* enc_wih = (const float*)d_in[8];
  const float* enc_whh = (const float*)d_in[9];
  const float* enc_bih = (const float*)d_in[10];
  const float* enc_bhh = (const float*)d_in[11];
  const float* dec_wih = (const float*)d_in[12];
  const float* dec_whh = (const float*)d_in[13];
  const float* dec_bih = (const float*)d_in[14];
  const float* dec_bhh = (const float*)d_in[15];
  const float* wq      = (const float*)d_in[16];
  const float* wk      = (const float*)d_in[17];
  const float* wv      = (const float*)d_in[18];
  const float* lnw     = (const float*)d_in[19];
  const float* lnb     = (const float*)d_in[20];
  float* out = (float*)d_out;

  char* p = (char*)d_ws;
  auto alloc = [&](size_t bytes) { char* r = p; p += (bytes + 255) & ~(size_t)255; return r; };

  _Float16* wenc = (_Float16*)alloc((size_t)2048 * 640 * 2);
  _Float16* wdec = (_Float16*)alloc((size_t)2048 * 1024 * 2);
  _Float16* wqh  = (_Float16*)alloc((size_t)HH * HH * 2);
  _Float16* wkh  = (_Float16*)alloc((size_t)HH * HH * 2);
  _Float16* wvh  = (_Float16*)alloc((size_t)HH * HH * 2);
  float*    bias_e = (float*)alloc(2048 * 4);
  float*    bias_d = (float*)alloc(2048 * 4);
  float*    wy     = (float*)alloc(2048 * 4);
  _Float16* feat = (_Float16*)alloc((size_t)BATCH * TT * 128 * 2);
  _Float16* xenc = (_Float16*)alloc((size_t)BATCH * TT * HH * 2);
  _Float16* Aenc = (_Float16*)alloc((size_t)BATCH * 640 * 2);
  _Float16* Adec = (_Float16*)alloc((size_t)BATCH * 1024 * 2);
  float*    gates = (float*)alloc((size_t)BATCH * 2048 * 4);   // reused as beta (f16) later
  float*    c_e  = (float*)alloc((size_t)BATCH * HH * 4);
  float*    c_d  = (float*)alloc((size_t)BATCH * HH * 4);
  _Float16* hd   = (_Float16*)alloc((size_t)BATCH * HH * 2);
  _Float16* qh   = (_Float16*)alloc((size_t)BATCH * HH * 2);
  _Float16* kh   = (_Float16*)alloc((size_t)BATCH * HH * 2);
  _Float16* vT   = (_Float16*)alloc((size_t)HH * BATCH * 2);
  float*    z    = (float*)alloc((size_t)BATCH * BATCH * 4);
  float*    st   = (float*)alloc((size_t)BATCH * HH * 4);
  _Float16* beta = (_Float16*)gates;                            // alias: 33.5MB each

  const dim3 blk(256);
  const dim3 ew(2048);

  // ---- weight / bias prep
  k_f32_to_f16<<<ew, blk, 0, stream>>>(wq, wqh, (long)HH * HH);
  k_f32_to_f16<<<ew, blk, 0, stream>>>(wk, wkh, (long)HH * HH);
  k_f32_to_f16<<<ew, blk, 0, stream>>>(wv, wvh, (long)HH * HH);
  k_build_w<<<ew, blk, 0, stream>>>(enc_wih, 128, 128, enc_whh, wenc, (long)2048 * 640);
  k_build_w<<<ew, blk, 0, stream>>>(dec_wih, 513, 512, dec_whh, wdec, (long)2048 * 1024);
  k_add_bias<<<dim3(8), blk, 0, stream>>>(enc_bih, enc_bhh, bias_e, 2048);
  k_add_bias<<<dim3(8), blk, 0, stream>>>(dec_bih, dec_bhh, bias_d, 2048);
  k_extract_col<<<dim3(8), blk, 0, stream>>>(dec_wih, 513, 512, wy, 2048);

  // ---- zero-init state / padded feat
  k_zero_f16<<<ew, blk, 0, stream>>>(feat, (long)BATCH * TT * 128);
  k_zero_f16<<<ew, blk, 0, stream>>>(Aenc, (long)BATCH * 640);
  k_zero_f16<<<ew, blk, 0, stream>>>(Adec, (long)BATCH * 1024);
  k_zero_f32<<<ew, blk, 0, stream>>>(c_e, (long)BATCH * HH);
  k_zero_f32<<<ew, blk, 0, stream>>>(c_d, (long)BATCH * HH);

  // ---- RCNN multi-scale features: (stride, Ls, chanoff)
  k_rcnn<<<dim3(BATCH), blk, 0, stream>>>(x, rw1 + 0 * 16 * 128, rb1 + 0 * 16, rw2 + 0 * 32 * 48, rb2 + 0 * 32, rw3 + 0 * 32 * 96, rb3 + 0 * 32, feat, 1, 45, 0);
  k_rcnn<<<dim3(BATCH), blk, 0, stream>>>(x, rw1 + 1 * 16 * 128, rb1 + 1 * 16, rw2 + 1 * 32 * 48, rb2 + 1 * 32, rw3 + 1 * 32 * 96, rb3 + 1 * 32, feat, 2, 23, 32);
  k_rcnn<<<dim3(BATCH), blk, 0, stream>>>(x, rw1 + 2 * 16 * 128, rb1 + 2 * 16, rw2 + 2 * 32 * 48, rb2 + 2 * 32, rw3 + 2 * 32 * 96, rb3 + 2 * 32, feat, 3, 15, 64);
  k_rcnn<<<dim3(BATCH), blk, 0, stream>>>(x, rw1 + 3 * 16 * 128, rb1 + 3 * 16, rw2 + 3 * 32 * 48, rb2 + 3 * 32, rw3 + 3 * 32 * 96, rb3 + 3 * 32, feat, 4, 12, 96);

  // ---- encoder LSTM: gates = [feat_t | h] @ [Wih|Whh]^T ; K=640
  for (int t = 0; t < TT; ++t) {
    k_copy_slice<<<ew, blk, 0, stream>>>(feat + (long)t * 128, (long)TT * 128, Aenc, 640, 128, (long)BATCH * 128);
    k_gemm<0><<<dim3(2048 / 256, BATCH / 128), blk, 0, stream>>>(Aenc, wenc, gates, BATCH, 2048, 640, 2048, 1.f);
    k_lstm_cell<<<dim3(BATCH * HH / 256), blk, 0, stream>>>(gates, bias_e, c_e, Aenc, 640, 128,
        xenc + (long)t * HH, (long)TT * HH, nullptr, nullptr, t);
  }

  // ---- decoder LSTM: gates = [xenc_t | h] @ W^T + y_t * wy ; K=1024
  for (int t = 0; t < TT; ++t) {
    k_copy_slice<<<ew, blk, 0, stream>>>(xenc + (long)t * HH, (long)TT * HH, Adec, 1024, HH, (long)BATCH * HH);
    k_gemm<0><<<dim3(2048 / 256, BATCH / 128), blk, 0, stream>>>(Adec, wdec, gates, BATCH, 2048, 1024, 2048, 1.f);
    k_lstm_cell<<<dim3(BATCH * HH / 256), blk, 0, stream>>>(gates, bias_d, c_d, Adec, 1024, 512,
        (t == TT - 1) ? hd : nullptr, (long)HH, y, wy, t);
  }

  // ---- attention
  k_gemm<1><<<dim3(HH / 256, BATCH / 128), blk, 0, stream>>>(hd, wqh, qh, BATCH, HH, HH, HH, 1.f);
  k_gemm<1><<<dim3(HH / 256, BATCH / 128), blk, 0, stream>>>(hd, wkh, kh, BATCH, HH, HH, HH, 1.f);
  k_gemm<2><<<dim3(HH / 256, BATCH / 128), blk, 0, stream>>>(hd, wvh, vT, BATCH, HH, HH, BATCH, 1.f);
  k_gemm<0><<<dim3(BATCH / 256, BATCH / 128), blk, 0, stream>>>(qh, kh, z, BATCH, BATCH, HH, BATCH, 0.04419417382415922f);
  k_softmax<<<dim3(BATCH), blk, 0, stream>>>(z, beta);
  k_gemm<0><<<dim3(HH / 256, BATCH / 128), blk, 0, stream>>>(beta, vT, st, BATCH, HH, BATCH, HH, 1.f);
  k_final<<<dim3(BATCH / 8), blk, 0, stream>>>(st, lnw, lnb, out);
}
